// Attention_75814762709205
// MI455X (gfx1250) — matrix operand order repaired and checked by reading, not compiled, not run on hardware
//
#include <hip/hip_runtime.h>
#include <hip/hip_bf16.h>
#include <stdint.h>

typedef __attribute__((ext_vector_type(16))) __bf16 v16bf;
typedef __attribute__((ext_vector_type(2)))  __bf16 v2bf;
typedef __attribute__((ext_vector_type(2)))  float  v2f;
typedef __attribute__((ext_vector_type(8)))  float  v8f;

#define Bn 32
#define Sn 2048
#define Dn 1024
#define TILE_M 64   // rows of (b,s) per workgroup

__device__ __forceinline__ __bf16 f2bf(float f) {
    union { float f; uint32_t u; } c; c.f = f;
    uint32_t u = c.u;
    uint32_t r = (u + 0x7FFFu + ((u >> 16) & 1u)) >> 16;   // round-to-nearest-even
    union { unsigned short s; __bf16 b; } o;
    o.s = (unsigned short)r;
    return o.b;
}

// Packed fp32x2 -> bf16x2: vector fptrunc, lets the backend pick the native
// packed convert if gfx1250 has one (falls back to legalized sequence if not).
__device__ __forceinline__ v2bf pk_bf16(float x, float y) {
    v2f p; p[0] = x; p[1] = y;
    return __builtin_convertvector(p, v2bf);
}

// ---------------------------------------------------------------------------
// Pack W (DxD fp32, row-major [k][n]) into WMMA-B-fragment-major bf16:
//   Wp[((kb*64 + ntile)*32 + lane)*16 + j]
// B k order used here (16-bit, 32x16, contiguous, not AMD's table order): VGPR v holds K = 2v + 16*half, 2v+1 + 16*half;
// column N = lane&15.  j = element index in v16bf (VGPR = j>>1, lo/hi = j&1).
// ---------------------------------------------------------------------------
__global__ void pack_w_kernel(const float* __restrict__ W,
                              unsigned short* __restrict__ Wp) {
    int idx = blockIdx.x * blockDim.x + threadIdx.x;     // 0 .. 2^20-1
    int j    = idx & 15;
    int lane = (idx >> 4) & 31;
    int nt   = (idx >> 9) & 63;
    int kb   = idx >> 15;
    int half = lane >> 4;
    int lnib = lane & 15;
    int kl   = 2 * (j >> 1) + (j & 1) + 16 * half;
    int k    = kb * 32 + kl;
    int n    = nt * 16 + lnib;
    union { __bf16 b; unsigned short s; } o;
    o.b = f2bf(W[(size_t)k * Dn + n]);
    Wp[idx] = o.s;
}

__global__ void zero_out_kernel(float* __restrict__ out, int n) {
    int i = blockIdx.x * blockDim.x + threadIdx.x;
    if (i < n) out[i] = 0.0f;
}

// ---------------------------------------------------------------------------
// Fused: proj = enc@W + b ; score = hd*proj ; softmax over D ; context += attn*enc
// Grid: (Sn/TILE_M, Bn), block: 512 threads = 16 waves (2 M-waves x 8 N-waves).
// Each wave: M = 32 rows (2 tiles), N = 128 cols (8 tiles), K-loop over 1024.
// ---------------------------------------------------------------------------
__global__ __launch_bounds__(512)
void attn_fused_kernel(const float* __restrict__ hd,
                       const float* __restrict__ enc,
                       const float* __restrict__ bias,
                       const unsigned short* __restrict__ wp,
                       float* __restrict__ out) {
    __shared__ float partial[TILE_M * 8];
    __shared__ float rowmax_s[TILE_M];
    __shared__ float rowsum_s[TILE_M];
    __shared__ float ctx[Dn];

    const int tid  = threadIdx.x;
    const int lane = tid & 31;
    const int wave = tid >> 5;
    const int wm   = wave >> 3;   // 0..1  : which 32-row block
    const int wn   = wave & 7;    // 0..7  : which 128-col block
    const int half = lane >> 4;
    const int lnib = lane & 15;
    const int b    = blockIdx.y;
    const int s0   = blockIdx.x * TILE_M;

    for (int i = tid; i < Dn; i += 512) ctx[i] = 0.0f;

    // per-lane column constants (score scale + bias)
    float hdv[8], bv[8];
#pragma unroll
    for (int nt = 0; nt < 8; ++nt) {
        int n = wn * 128 + nt * 16 + lnib;
        hdv[nt] = hd[n];
        bv[nt]  = bias[n];
    }

    // A-fragment row pointers (A layout: row M = lane&15, both halves)
    const float* row0 = enc + ((size_t)b * Sn + (s0 + wm * 32 + lnib)) * Dn;      // mt=0
    const float* row1 = row0 + (size_t)16 * Dn;                                    // mt=1

    v8f zero = {};
    v8f acc[2][8];
#pragma unroll
    for (int mt = 0; mt < 2; ++mt)
#pragma unroll
        for (int nt = 0; nt < 8; ++nt) acc[mt][nt] = zero;

    // ---- GEMM: proj = enc @ W -------------------------------------------------
    for (int k0 = 0; k0 < Dn; k0 += 32) {
        v16bf a0, a1;
#pragma unroll
        for (int v = 0; v < 8; ++v) {
            // A k order used here (contiguous, matches packed B): VGPR v -> K = 2*v + 16*half
            int kl = 2 * v + 16 * half;
            float2 p0 = *(const float2*)(row0 + k0 + kl);
            float2 p1 = *(const float2*)(row1 + k0 + kl);
            v2bf c0 = pk_bf16(p0.x, p0.y);
            v2bf c1 = pk_bf16(p1.x, p1.y);
            a0[2 * v]     = c0[0];
            a0[2 * v + 1] = c0[1];
            a1[2 * v]     = c1[0];
            a1[2 * v + 1] = c1[1];
        }
        const unsigned short* wb =
            wp + ((size_t)((k0 >> 5) * 64 + wn * 8) * 32 + lane) * 16;
#pragma unroll
        for (int nt = 0; nt < 8; ++nt) {
            v16bf bf = *(const v16bf*)(wb + (size_t)nt * (32 * 16));
            acc[0][nt] = __builtin_amdgcn_wmma_f32_16x16x32_bf16(
                false, a0, false, bf, (short)0, acc[0][nt], false, false);
            acc[1][nt] = __builtin_amdgcn_wmma_f32_16x16x32_bf16(
                false, a1, false, bf, (short)0, acc[1][nt], false, false);
        }
    }

    // ---- score = hd * (proj + bias) ------------------------------------------
#pragma unroll
    for (int mt = 0; mt < 2; ++mt)
#pragma unroll
        for (int nt = 0; nt < 8; ++nt)
#pragma unroll
            for (int v = 0; v < 8; ++v)
                acc[mt][nt][v] = hdv[nt] * (acc[mt][nt][v] + bv[nt]);

    // ---- row max (C layout: col = lnib, row = v + 8*half within tile) --------
#pragma unroll
    for (int mt = 0; mt < 2; ++mt) {
#pragma unroll
        for (int v = 0; v < 8; ++v) {
            float m = acc[mt][0][v];
#pragma unroll
            for (int nt = 1; nt < 8; ++nt) m = fmaxf(m, acc[mt][nt][v]);
#pragma unroll
            for (int off = 1; off < 16; off <<= 1)
                m = fmaxf(m, __shfl_xor(m, off, 32));
            if (lnib == 0)
                partial[(wm * 32 + mt * 16 + 8 * half + v) * 8 + wn] = m;
        }
    }
    __syncthreads();
    if (tid < TILE_M) {
        float m = partial[tid * 8];
#pragma unroll
        for (int j = 1; j < 8; ++j) m = fmaxf(m, partial[tid * 8 + j]);
        rowmax_s[tid] = m;
    }
    __syncthreads();

    // ---- exp + row sum --------------------------------------------------------
#pragma unroll
    for (int mt = 0; mt < 2; ++mt) {
#pragma unroll
        for (int v = 0; v < 8; ++v) {
            float rm = rowmax_s[wm * 32 + mt * 16 + 8 * half + v];
            float s = 0.0f;
#pragma unroll
            for (int nt = 0; nt < 8; ++nt) {
                float e = __expf(acc[mt][nt][v] - rm);
                acc[mt][nt][v] = e;
                s += e;
            }
#pragma unroll
            for (int off = 1; off < 16; off <<= 1)
                s += __shfl_xor(s, off, 32);
            if (lnib == 0)
                partial[(wm * 32 + mt * 16 + 8 * half + v) * 8 + wn] = s;
        }
    }
    __syncthreads();
    if (tid < TILE_M) {
        float s = 0.0f;
#pragma unroll
        for (int j = 0; j < 8; ++j) s += partial[tid * 8 + j];
        rowsum_s[tid] = s;
    }
    __syncthreads();

    float inv[2][8];
#pragma unroll
    for (int mt = 0; mt < 2; ++mt)
#pragma unroll
        for (int v = 0; v < 8; ++v)
            inv[mt][v] = 1.0f / rowsum_s[wm * 32 + mt * 16 + 8 * half + v];

    // ---- context: sum_s attn * enc -------------------------------------------
    const float* erow = enc + ((size_t)b * Sn + (s0 + wm * 32 + 8 * half)) * Dn;
#pragma unroll
    for (int nt = 0; nt < 8; ++nt) {
        int n = wn * 128 + nt * 16 + lnib;
        float cs = 0.0f;
#pragma unroll
        for (int v = 0; v < 8; ++v) {
            cs += acc[0][nt][v] * inv[0][v] * erow[(size_t)v * Dn + n];
            cs += acc[1][nt][v] * inv[1][v] * erow[(size_t)(16 + v) * Dn + n];
        }
        cs += __shfl_xor(cs, 16, 32);       // combine the two halves (rows 0-7 / 8-15)
        if (half == 0) atomicAdd(&ctx[n], cs);
    }
    __syncthreads();

    for (int i = tid; i < Dn; i += 512)
        atomicAdd(&out[(size_t)b * Dn + i], ctx[i]);
}

// ---------------------------------------------------------------------------
extern "C" void kernel_launch(void* const* d_in, const int* in_sizes, int n_in,
                              void* d_out, int out_size, void* d_ws, size_t ws_size,
                              hipStream_t stream) {
    (void)in_sizes; (void)n_in; (void)out_size; (void)ws_size;
    const float* hd   = (const float*)d_in[0];   // [D,1]
    const float* enc  = (const float*)d_in[1];   // [B,S,D]
    const float* W    = (const float*)d_in[2];   // [D,D]
    const float* bias = (const float*)d_in[3];   // [D]
    float* out = (float*)d_out;                  // [B,D]
    unsigned short* wp = (unsigned short*)d_ws;  // 2 MB packed bf16 W

    zero_out_kernel<<<(Bn * Dn + 255) / 256, 256, 0, stream>>>(out, Bn * Dn);
    pack_w_kernel<<<(Dn * Dn) / 256, 256, 0, stream>>>(W, wp);
    dim3 grid(Sn / TILE_M, Bn);
    attn_fused_kernel<<<grid, 512, 0, stream>>>(hd, enc, bias, wp, out);
}
